// MiniKPCNNSingleFrame_28767690949426
// MI455X (gfx1250) — compile-verified, matrix operand order checked by reading, not run on hardware
//
#include <hip/hip_runtime.h>
#include <hip/hip_bf16.h>
#include <stdint.h>

// ---------------------------------------------------------------------------
// CDNA5 WMMA types
// ---------------------------------------------------------------------------
typedef __attribute__((ext_vector_type(16))) __bf16    v16bf;
typedef __attribute__((ext_vector_type(8)))  float     v8f;
typedef __attribute__((ext_vector_type(4)))  uint32_t  u32x4;

__device__ __forceinline__ unsigned short f32_to_bf16(float f) {
  union { float f; uint32_t u; } c; c.f = f;
  uint32_t u = c.u;
  uint32_t r = (u + 0x7FFFu + ((u >> 16) & 1u)) >> 16;   // round-nearest-even
  return (unsigned short)r;
}

__device__ __forceinline__ int reflect_idx(int i, int n) {
  if (i < 0) i = -i;
  if (i >= n) i = 2 * n - 2 - i;
  return i;
}

// ---------------------------------------------------------------------------
// One-shot weight pack: f32 (C_OUT,C_IN,KH,KW) -> bf16 wp[ROWS][K_PAD] with
// channel-fastest K (k = tap*C_IN_PAD + cin), zero padded in cin/k/rows.
// ---------------------------------------------------------------------------
template <int C_IN, int C_IN_PAD, int ROWS, int C_OUT, int KH, int KW>
__global__ __launch_bounds__(256) void pack_weights_kernel(
    const float* __restrict__ wgt, unsigned short* __restrict__ wp) {
  constexpr int K_EFF = KH * KW * C_IN_PAD;
  constexpr int K_PAD = (K_EFF + 31) & ~31;
  int idx = blockIdx.x * 256 + threadIdx.x;
  if (idx >= ROWS * K_PAD) return;
  int c = idx / K_PAD, k = idx % K_PAD;
  unsigned short v = 0;
  if (k < K_EFF && c < C_OUT) {
    int tap = k / C_IN_PAD, cin = k % C_IN_PAD;
    if (cin < C_IN) {
      int ky = tap / KW, kx = tap % KW;
      v = f32_to_bf16(wgt[(((size_t)c * C_IN + cin) * KH + ky) * KW + kx]);
    }
  }
  wp[idx] = v;
}

// ---------------------------------------------------------------------------
// Implicit-GEMM conv, reflect padding, bf16 WMMA, f32 accumulate.
//   K order: k = (ky*KW+kx)*C_IN_PAD + cin,  C_IN_PAD % 16 == 0
//   => hiK folds into base pointers; fully-unrolled K loop makes every
//      fragment offset an immediate on ds_load_b128 (pure DS + WMMA loop).
//   Register blocking: each wave computes CPV(16|32) channels x 32 pixels:
//      AR=CPV/16 A-frags x 2 B-frags -> AR*2 WMMAs per chunk
//      (CPV=32: 8 b128 loads / 4 WMMAs = 2.0 loads per WMMA).
//   WG: CW channel wave-rows x PW pixel wave-cols. Activations NHWC global.
// ---------------------------------------------------------------------------
template <int TPX, int CW, int CPV, int C_IN, int C_IN_PAD, int C_OUT,
          int KH, int KW, bool IN_NCHW_F32, bool OUT_F32, bool RELU>
__global__ __launch_bounds__(32 * CW * (TPX / 32)) void conv_wmma_kernel(
    const void* __restrict__ src,            // NCHW f32 (layer1) or NHWC bf16
    const unsigned short* __restrict__ wp,   // packed bf16 weights
    const float* __restrict__ bias,
    void* __restrict__ dst,                  // NHWC: bf16 or f32
    int Wimg, int Himg) {
  constexpr int PW    = TPX / 32;
  constexpr int NT    = 32 * CW * PW;
  constexpr int AR    = CPV / 16;            // A-fragment rows per wave
  constexpr int CT    = CW * CPV;
  constexpr int PAD   = KH / 2;
  constexpr int TW    = TPX + KW - 1;
  constexpr int K_EFF = KH * KW * C_IN_PAD;
  constexpr int K_PAD = (K_EFF + 31) & ~31;
  static_assert(C_IN_PAD % 16 == 0 && C_IN_PAD >= C_IN, "pad");
  static_assert(C_OUT % 8 == 0, "store runs");
  static_assert(CPV == 16 || CPV == 32, "wave channel tile");

  __shared__ unsigned short lds_in[KH * TW * C_IN_PAD];
  __shared__ unsigned short lds_w[CT * K_PAD];

  const int tid    = threadIdx.x;
  const int tilesx = Wimg / TPX;
  const int ty     = blockIdx.x / tilesx;
  const int x0     = (blockIdx.x % tilesx) * TPX;
  const int cbase  = blockIdx.y * CT;
  const int b      = blockIdx.z;
  const size_t plane = (size_t)Wimg * (size_t)Himg;

  // --- stage packed weights: straight vector copy --------------------------
  {
    const u32x4* s = (const u32x4*)(wp + (size_t)cbase * K_PAD);
    u32x4*       d = (u32x4*)lds_w;
    for (int i = tid; i < CT * K_PAD / 8; i += NT) d[i] = s[i];
  }

  // --- stage reflect-padded input strip ------------------------------------
  if (IN_NCHW_F32) {
    const float* srcF = (const float*)src;
    for (int i = tid; i < C_IN * KH * TW; i += NT) {    // read-coalesced order
      int col = i % TW;
      int rr  = i / TW;
      int ky  = rr % KH;
      int cin = rr / KH;
      int iy  = reflect_idx(ty + ky - PAD, Himg);
      int ix  = reflect_idx(x0 + col - PAD, Wimg);
      lds_in[(ky * TW + col) * C_IN_PAD + cin] =
          f32_to_bf16(srcF[((size_t)b * C_IN + cin) * plane +
                           (size_t)iy * Wimg + ix]);
    }
    if (C_IN_PAD > C_IN) {
      constexpr int PC = C_IN_PAD - C_IN;
      for (int i = tid; i < KH * TW * PC; i += NT)
        lds_in[(i / PC) * C_IN_PAD + C_IN + (i % PC)] = 0;
    }
  } else {
    constexpr int CHUNKS = C_IN / 8;                    // 16B chunks per run
    const unsigned short* srcH = (const unsigned short*)src;
    for (int id = tid; id < KH * TW * CHUNKS; id += NT) {
      int part = id % CHUNKS;
      int run  = id / CHUNKS;
      int col  = run % TW;
      int ky   = run / TW;
      int iy   = reflect_idx(ty + ky - PAD, Himg);
      int ix   = reflect_idx(x0 + col - PAD, Wimg);
      const u32x4* g = (const u32x4*)(srcH +
          ((size_t)(b * Himg + iy) * Wimg + ix) * C_IN + part * 8);
      *(u32x4*)&lds_in[run * C_IN_PAD + part * 8] = *g;
    }
  }

  // prefetch next strip's first pixel row (global_prefetch_b8 path)
  if (tid < 16) {
    size_t g = IN_NCHW_F32
        ? (((size_t)b * C_IN + tid) * plane + (size_t)ty * Wimg +
           ((x0 + TPX < Wimg) ? (x0 + TPX) : x0)) * 4
        : (((size_t)(b * Himg + ty) * Wimg +
            ((x0 + TPX < Wimg) ? (x0 + TPX) : x0)) * C_IN + tid * 8) * 2;
    __builtin_prefetch((const char*)src + g, 0, 3);
  }

  __syncthreads();

  // --- per-wave CPV(ch) x 32(px) WMMA tiles --------------------------------
  const int wave  = tid >> 5;
  const int lane  = tid & 31;
  const int cw    = wave / PW;              // 0..CW-1
  const int pw    = wave % PW;              // 0..PW-1
  const int mn    = lane & 15;
  const int hiK   = (lane >= 16) ? 8 : 0;
  const int pbase = pw * 32;
  const int clo   = cw * CPV;

  union Frag { v16bf v; u32x4 q[2]; };
  v8f acc[AR][2];
#pragma unroll
  for (int r = 0; r < AR; ++r)
#pragma unroll
    for (int t = 0; t < 2; ++t) acc[r][t] = (v8f){};

  const unsigned short* aptr[AR];
#pragma unroll
  for (int r = 0; r < AR; ++r)
    aptr[r] = &lds_w[(clo + r * 16 + mn) * K_PAD + hiK];
  const unsigned short* bp0 = &lds_in[(pbase + mn) * C_IN_PAD + hiK];
  const unsigned short* bp1 = bp0 + 16 * C_IN_PAD;

#pragma unroll
  for (int k0 = 0; k0 < K_PAD; k0 += 32) {
    Frag a[AR], bf[2];
#pragma unroll
    for (int half = 0; half < 2; ++half) {
      const int kb = k0 + 16 * half;                // compile-time
#pragma unroll
      for (int r = 0; r < AR; ++r)
        a[r].q[half] = *(const u32x4*)(aptr[r] + kb);
      const int kc  = (kb < K_EFF) ? kb : 0;        // clamped pad chunk
      const int tap = kc / C_IN_PAD;                // C_IN_PAD%16==0 => hiK
      const int cin = kc % C_IN_PAD;                //   stays inside the tap
      const int off = ((tap / KW) * TW + (tap % KW)) * C_IN_PAD + cin;
      bf[0].q[half] = *(const u32x4*)(bp0 + off);
      bf[1].q[half] = *(const u32x4*)(bp1 + off);
    }
#pragma unroll
    for (int r = 0; r < AR; ++r)
#pragma unroll
      for (int t = 0; t < 2; ++t)
        acc[r][t] = __builtin_amdgcn_wmma_f32_16x16x32_bf16(
            false, a[r].v, false, bf[t].v, (short)0, acc[r][t], false, false);
  }

  // --- store: lane holds 8 consecutive channels per pixel (NHWC) -----------
#pragma unroll
  for (int r = 0; r < AR; ++r) {
    const int c0 = cbase + clo + r * 16 + hiK;
    if (c0 + 8 <= C_OUT) {               // C_OUT % 8 == 0: run all-or-nothing
#pragma unroll
      for (int t = 0; t < 2; ++t) {
        const int px = x0 + pbase + t * 16 + mn;
        size_t base = ((size_t)(b * Himg + ty) * Wimg + px) * C_OUT + c0;
        float v[8];
#pragma unroll
        for (int e = 0; e < 8; ++e) {
          v[e] = acc[r][t][e] + bias[c0 + e];
          if (RELU) v[e] = fmaxf(v[e], 0.0f);
        }
        if (OUT_F32) {
          float* dstF = (float*)dst;
#pragma unroll
          for (int e = 0; e < 8; ++e) dstF[base + e] = v[e]; // 32B contiguous
        } else {
          u32x4 packed;
#pragma unroll
          for (int e = 0; e < 4; ++e)
            packed[e] = (uint32_t)f32_to_bf16(v[2 * e]) |
                        ((uint32_t)f32_to_bf16(v[2 * e + 1]) << 16);
          *(u32x4*)((unsigned short*)dst + base) = packed;   // one b128 store
        }
      }
    }
  }
}

// ---------------------------------------------------------------------------
// Head: softmax over 16 logits (NHWC raw), fold einsum into an effective
// 9x9 tap kernel: filtered[c] = sum_t color[c,tap] * (sum_k w_k*dwk[k,t]).
// Output d_out is NCHW f32 (B, 11, H, W): ch 0..2 filtered, 3..10 hidden.
// ---------------------------------------------------------------------------
__global__ __launch_bounds__(256) void kpn_head_kernel(
    const float* __restrict__ x,    // (B, 20, H, W) NCHW, color = ch 0..2
    const float* __restrict__ raw,  // (B, H, W, 24) NHWC
    const float* __restrict__ dwk,  // (16, 1, 9, 9)
    float* __restrict__ out,        // (B, 11, H, W) NCHW
    int Wimg, int Himg, int B) {
  constexpr int KK = 16, P = 9, TAPS = P * P, HID = 8;
  __shared__ float lds_dwk[TAPS * KK];   // transposed: [tap][k]

  for (int i = threadIdx.x; i < TAPS * KK; i += 256) {
    int t = i / KK, k = i % KK;
    lds_dwk[i] = dwk[k * TAPS + t];
  }
  __syncthreads();

  const size_t plane = (size_t)Wimg * (size_t)Himg;
  size_t gid = (size_t)blockIdx.x * 256 + threadIdx.x;
  if (gid >= (size_t)B * plane) return;
  const int b  = (int)(gid / plane);
  const size_t p = gid % plane;
  const int y  = (int)(p / Wimg);
  const int xx = (int)(p % Wimg);

  const size_t rbase = gid * 24;         // NHWC pixel record

  float w[KK];
  float mx = -3.0e38f;
#pragma unroll
  for (int k = 0; k < KK; ++k) {
    w[k] = raw[rbase + HID + k] * 0.2f;  // contiguous 64B of logits
    mx = fmaxf(mx, w[k]);
  }
  float s = 0.0f;
#pragma unroll
  for (int k = 0; k < KK; ++k) { w[k] = __expf(w[k] - mx); s += w[k]; }
  float inv = 1.0f / s;
#pragma unroll
  for (int k = 0; k < KK; ++k) w[k] *= inv;

  float a0 = 0.0f, a1 = 0.0f, a2 = 0.0f;
  for (int t = 0; t < TAPS; ++t) {
    int iy = y + t / P - P / 2;
    int ix = xx + t % P - P / 2;
    if ((unsigned)iy < (unsigned)Himg && (unsigned)ix < (unsigned)Wimg) {
      float wk = 0.0f;
#pragma unroll
      for (int k = 0; k < KK; ++k) wk = fmaf(w[k], lds_dwk[t * KK + k], wk);
      size_t ip = (size_t)iy * Wimg + ix;
      a0 = fmaf(x[((size_t)b * 20 + 0) * plane + ip], wk, a0);
      a1 = fmaf(x[((size_t)b * 20 + 1) * plane + ip], wk, a1);
      a2 = fmaf(x[((size_t)b * 20 + 2) * plane + ip], wk, a2);
    }
  }

  out[((size_t)b * 11 + 0) * plane + p] = a0;
  out[((size_t)b * 11 + 1) * plane + p] = a1;
  out[((size_t)b * 11 + 2) * plane + p] = a2;
#pragma unroll
  for (int h = 0; h < HID; ++h)
    out[((size_t)b * 11 + 3 + h) * plane + p] = raw[rbase + h];
}

// ---------------------------------------------------------------------------
// Launch
// ---------------------------------------------------------------------------
extern "C" void kernel_launch(void* const* d_in, const int* in_sizes, int n_in,
                              void* d_out, int out_size, void* d_ws, size_t ws_size,
                              hipStream_t stream) {
  (void)in_sizes; (void)n_in; (void)out_size; (void)ws_size;
  const int Wimg = 1024, Himg = 1024, B = 2;
  const size_t plane = (size_t)Wimg * Himg;

  const float* x   = (const float*)d_in[0];
  const float* w1  = (const float*)d_in[1];
  const float* b1  = (const float*)d_in[2];
  const float* w2  = (const float*)d_in[3];
  const float* b2  = (const float*)d_in[4];
  const float* w3  = (const float*)d_in[5];
  const float* b3  = (const float*)d_in[6];
  const float* w4  = (const float*)d_in[7];
  const float* b4  = (const float*)d_in[8];
  const float* dwk = (const float*)d_in[9];

  // workspace: NHWC activations (bf16 h1..h3, f32 raw) + packed weights
  char* ws = (char*)d_ws;
  unsigned short* h1  = (unsigned short*)(ws);                     //  64 MB
  unsigned short* h2  = (unsigned short*)(ws + (size_t)67108864);  // 128 MB
  unsigned short* h3  = (unsigned short*)(ws + (size_t)201326592); // 256 MB
  float*          raw = (float*)        (ws + (size_t)469762048);  // 192 MB
  unsigned short* wp1 = (unsigned short*)(ws + (size_t)671088640);
  unsigned short* wp2 = (unsigned short*)(ws + (size_t)671088640 + 262144);
  unsigned short* wp3 = (unsigned short*)(ws + (size_t)671088640 + 524288);
  unsigned short* wp4 = (unsigned short*)(ws + (size_t)671088640 + 786432);

  // ---- pack weights (K_PAD: L1 49*32=1568, L2 416, L3 800, L4 1600) -------
  pack_weights_kernel<20, 32, 16, 16, 7, 7>
      <<<dim3((16 * 1568 + 255) / 256), 256, 0, stream>>>(w1, wp1);
  pack_weights_kernel<16, 16, 32, 32, 5, 5>
      <<<dim3((32 * 416 + 255) / 256), 256, 0, stream>>>(w2, wp2);
  pack_weights_kernel<32, 32, 64, 64, 5, 5>
      <<<dim3((64 * 800 + 255) / 256), 256, 0, stream>>>(w3, wp3);
  pack_weights_kernel<64, 64, 32, 24, 5, 5>
      <<<dim3((32 * 1600 + 255) / 256), 256, 0, stream>>>(w4, wp4);

  // ---- convolutions -------------------------------------------------------
  const int tiles128 = (Wimg / 128) * Himg;   // 8192
  const int tiles64  = (Wimg / 64)  * Himg;   // 16384

  // <TPX, CW, CPV, C_IN, C_IN_PAD, C_OUT, KH, KW, IN_NCHW_F32, OUT_F32, RELU>
  conv_wmma_kernel<128, 1, 16, 20, 32, 16, 7, 7, true,  false, true >
      <<<dim3(tiles128, 1, B), 128, 0, stream>>>(x,  wp1, b1, h1,  Wimg, Himg);
  conv_wmma_kernel<128, 1, 32, 16, 16, 32, 5, 5, false, false, true >
      <<<dim3(tiles128, 1, B), 128, 0, stream>>>(h1, wp2, b2, h2,  Wimg, Himg);
  conv_wmma_kernel<128, 2, 32, 32, 32, 64, 5, 5, false, false, true >
      <<<dim3(tiles128, 1, B), 256, 0, stream>>>(h2, wp3, b3, h3,  Wimg, Himg);
  conv_wmma_kernel< 64, 1, 32, 64, 64, 24, 5, 5, false, true,  false>
      <<<dim3(tiles64,  1, B),  64, 0, stream>>>(h3, wp4, b4, raw, Wimg, Himg);

  // ---- head ---------------------------------------------------------------
  const size_t npix = (size_t)B * plane;
  kpn_head_kernel<<<dim3((unsigned)((npix + 255) / 256)), 256, 0, stream>>>(
      x, raw, dwk, (float*)d_out, Wimg, Himg, B);
}
